// SparseSVDFusedWrap_62706522522286
// MI455X (gfx1250) — matrix operand order repaired and checked by reading, not compiled, not run on hardware
//
#include <hip/hip_runtime.h>

// ---------------------------------------------------------------------------
// Problem constants (from the reference)
// ---------------------------------------------------------------------------
#define D_IN    4096
#define D_OUT   4096
#define RANK    128
#define M_TOTAL 16384        // BATCH * SEQ = 4 * 4096

// GEMM tiling: block = 256 threads = 8 wave32, tile 128(M) x 128(N) x 32(K).
// Each wave owns a 32x64 patch = 2x4 WMMA tiles -> 24 v_wmma per K-step.
#define BM 128
#define BN 128
#define BK 32
#define LDS_STRIDE 40        // padded row stride in bf16 elems (80 bytes)

// dynamic LDS: sAh | sAl | sBh | sBl, each double-buffered [2][rows][40] bf16
#define A_ELEMS (2 * BM * LDS_STRIDE)        // 10240 elems = 20480 B
#define B_ELEMS (2 * BN * LDS_STRIDE)        // 10240 elems = 20480 B
#define SMEM_BYTES ((2 * A_ELEMS + 2 * B_ELEMS) * 2)   // 81920 B < 320 KB WGP LDS

typedef __attribute__((ext_vector_type(16))) __bf16 v16bf;
typedef __attribute__((ext_vector_type(8)))  __bf16 bfx8;
typedef __attribute__((ext_vector_type(4)))  __bf16 bfx4;
typedef __attribute__((ext_vector_type(8)))  float  v8f;
typedef unsigned int uint32_t_;
typedef __attribute__((ext_vector_type(4)))  unsigned int v4u;
typedef __attribute__((ext_vector_type(8)))  unsigned int v8u;

// ---------------------------------------------------------------------------
// Kernel 1: fold the int8 low-rank correction into the dense weight and emit
// a split-bf16 (hi + lo) copy of W_eff in [o][i] (row = output feature) order.
//   W_eff[o][i] = weight[o][i] + (V_scale*U_T_scale) * sum_r V_q[i][r]*U_T_q[r][o]
// ---------------------------------------------------------------------------
__global__ void __launch_bounds__(256)
weff_build_kernel(const float* __restrict__ weight,
                  const int*   __restrict__ Vq,     // [D_IN][RANK]
                  const int*   __restrict__ UTq,    // [RANK][D_OUT]
                  const float* __restrict__ vscale,
                  const float* __restrict__ uscale,
                  __bf16* __restrict__ whi,         // [D_OUT][D_IN]
                  __bf16* __restrict__ wlo)         // [D_OUT][D_IN]
{
    __shared__ int vrow[RANK];
    const int i = blockIdx.y;                       // input feature (K index)
    const int o = blockIdx.x * 256 + threadIdx.x;   // output feature (N index)

    if (threadIdx.x < RANK) vrow[threadIdx.x] = Vq[i * RANK + threadIdx.x];
    __syncthreads();

    int acc = 0;
#pragma unroll 8
    for (int r = 0; r < RANK; ++r)
        acc += vrow[r] * UTq[r * D_OUT + o];        // coalesced across o, L2-resident

    const float w = weight[(size_t)o * D_IN + i] +
                    (float)acc * vscale[0] * uscale[0];
    const __bf16 h = (__bf16)w;
    const __bf16 l = (__bf16)(w - (float)h);
    whi[(size_t)o * D_IN + i] = h;
    wlo[(size_t)o * D_IN + i] = l;
}

// ---------------------------------------------------------------------------
// TDM issue: DMA one [BN rows x 32 bf16] W_eff tile (row stride D_IN) from
// global into LDS with hardware padding: pad_interval = 16 dwords (one 64 B
// row), pad_amount = 4 dwords (16 B) -> LDS row stride 80 B == LDS_STRIDE.
// D# layout per CDNA5 ISA 8.3/8.4 (2D tensor: groups 2/3 omitted -> NULL).
// ---------------------------------------------------------------------------
__device__ __forceinline__ void tdm_issue_b_tile(const __bf16* gsrc,
                                                 unsigned lds_byte_addr)
{
    const unsigned long long ga = (unsigned long long)(uintptr_t)gsrc;
    v4u g0;
    g0[0] = 1u;                                     // count=1, user descriptor
    g0[1] = lds_byte_addr;                          // LDS destination (bytes)
    g0[2] = (unsigned)ga;                           // global_addr[31:0]
    g0[3] = (unsigned)((ga >> 32) & 0x01FFFFFFu)    // global_addr[56:32]
            | (2u << 30);                           // type = 2 ("image")
    v8u g1;
    g1[0] = (1u << 16)                              // data_size = 1 -> 2 bytes
          | (1u << 20)                              // pad_enable
          | (3u << 22)                              // pad_interval: 16 dwords
          | (3u << 25);                             // pad_amount:   4 dwords
    g1[1] = (unsigned)(D_IN & 0xFFFF) << 16;        // tensor_dim0[15:0]
    g1[2] = ((unsigned)D_IN >> 16)                  // tensor_dim0[31:16]
          | ((unsigned)(D_OUT & 0xFFFF) << 16);     // tensor_dim1[15:0]
    g1[3] = ((unsigned)D_OUT >> 16)                 // tensor_dim1[31:16]
          | (32u << 16);                            // tile_dim0 = 32 elems
    g1[4] = (unsigned)BN;                           // tile_dim1 = 128, tile_dim2 = 0 (2D)
    g1[5] = (unsigned)D_IN;                         // tensor_dim0_stride[31:0]
    g1[6] = 0u;                                     // stride hi / dim1_stride lo
    g1[7] = 0u;
    asm volatile("tensor_load_to_lds %0, %1" :: "s"(g0), "s"(g1) : "memory");
}

// ---------------------------------------------------------------------------
// Fragment loaders (wave32 WMMA VGPR layouts, CDNA5 ISA 7.12.2)
// ---------------------------------------------------------------------------
__device__ __forceinline__ v16bf load_a_frag(const __bf16* base, int row, int hh)
{
    const bfx8 u = *(const bfx8*)(base + row * LDS_STRIDE + 8 * hh);       // K 0..7 / 8..15
    const bfx8 v = *(const bfx8*)(base + row * LDS_STRIDE + 16 + 8 * hh);  // K 16..23 / 24..31
    v16bf r;
#pragma unroll
    for (int e = 0; e < 8; ++e) { r[e] = u[e]; r[e + 8] = v[e]; }
    return r;
}

__device__ __forceinline__ v16bf load_b_frag(const __bf16* base, int col, int hh)
{
    const bfx8 u = *(const bfx8*)(base + col * LDS_STRIDE + 8 * hh);
    const bfx8 v = *(const bfx8*)(base + col * LDS_STRIDE + 16 + 8 * hh);
    v16bf r;
#pragma unroll
    for (int e = 0; e < 8; ++e) { r[e] = u[e]; r[e + 8] = v[e]; }
    return r;
}

// ---------------------------------------------------------------------------
// Kernel 2: C[M,N] = X[M,K] * W_eff[K,N] + bias, split-bf16 triple-WMMA,
// double-buffered LDS; B tiles DMA'd by the Tensor Data Mover, X tiles
// converted fp32 -> bf16(hi,lo) through the vector pipe.
// ---------------------------------------------------------------------------
__global__ void __launch_bounds__(256)
lora_gemm_wmma_kernel(const float*  __restrict__ X,    // [M_TOTAL][D_IN]
                      const __bf16* __restrict__ wh,   // [D_OUT][D_IN]
                      const __bf16* __restrict__ wl,   // [D_OUT][D_IN]
                      const float*  __restrict__ bias, // [D_OUT]
                      float*        __restrict__ out)  // [M_TOTAL][D_OUT]
{
    extern __shared__ __align__(16) __bf16 smem[];
    __bf16* sAh = smem;                  // [2][BM][LDS_STRIDE]
    __bf16* sAl = sAh + A_ELEMS;
    __bf16* sBh = sAl + A_ELEMS;         // [2][BN][LDS_STRIDE]
    __bf16* sBl = sBh + B_ELEMS;

    const int tid    = threadIdx.x;
    const int lane   = tid & 31;
    const int lane16 = lane & 15;
    const int hh     = (lane >> 4) & 1;
    const int wave   = tid >> 5;
    const int wm     = (wave & 3) * 32;   // 4 waves along M  -> 128
    const int wn     = (wave >> 2) * 64;  // 2 waves along N  -> 128

    const int m0 = blockIdx.y * BM;
    const int n0 = blockIdx.x * BN;

    // X-tile staging assignment: 4 x float4 per thread
    const int xr_row = tid >> 3;          // 0..31 (+32 per kk)
    const int xr_c4  = (tid & 7) * 4;     // float column within the 32-wide K tile

    v8f acc[2][4] = {};
    float4 xr[4];

    // ---- prologue: TDM B-tile 0, vector-load X-tile 0, stage into buffer 0
    if (wave == 0) {
        tdm_issue_b_tile(&wh[(size_t)n0 * D_IN], (unsigned)(uintptr_t)&sBh[0]);
        tdm_issue_b_tile(&wl[(size_t)n0 * D_IN], (unsigned)(uintptr_t)&sBl[0]);
    }
#pragma unroll
    for (int kk = 0; kk < 4; ++kk)
        xr[kk] = *(const float4*)&X[(size_t)(m0 + kk * 32 + xr_row) * D_IN + xr_c4];
#pragma unroll
    for (int kk = 0; kk < 4; ++kk) {
        const float f[4] = { xr[kk].x, xr[kk].y, xr[kk].z, xr[kk].w };
        bfx4 h4, l4;
#pragma unroll
        for (int e = 0; e < 4; ++e) {
            h4[e] = (__bf16)f[e];
            l4[e] = (__bf16)(f[e] - (float)h4[e]);
        }
        *(bfx4*)&sAh[(kk * 32 + xr_row) * LDS_STRIDE + xr_c4] = h4;
        *(bfx4*)&sAl[(kk * 32 + xr_row) * LDS_STRIDE + xr_c4] = l4;
    }
    __builtin_amdgcn_s_wait_tensorcnt(0);
    __syncthreads();

    const int KT = D_IN / BK;   // 128 K-steps
    for (int kt = 0; kt < KT; ++kt) {
        const int buf = kt & 1;
        const int nb  = buf ^ 1;

        // ---- issue next tile's TDM + global loads (overlap with WMMA) ----
        if (kt + 1 < KT) {
            const int kb = (kt + 1) * BK;
            if (wave == 0) {
                tdm_issue_b_tile(&wh[(size_t)n0 * D_IN + kb],
                                 (unsigned)(uintptr_t)&sBh[nb * BN * LDS_STRIDE]);
                tdm_issue_b_tile(&wl[(size_t)n0 * D_IN + kb],
                                 (unsigned)(uintptr_t)&sBl[nb * BN * LDS_STRIDE]);
            }
#pragma unroll
            for (int kk = 0; kk < 4; ++kk)
                xr[kk] = *(const float4*)&X[(size_t)(m0 + kk * 32 + xr_row) * D_IN + kb + xr_c4];
        }
        if (kt + 2 < KT) {
            // gfx1250 global_prefetch_b8: warm caches two K-tiles ahead
            __builtin_prefetch(&X[(size_t)(m0 + xr_row) * D_IN + (kt + 2) * BK], 0, 3);
        }

        // ---- WMMA compute on the current buffer ----
        const __bf16* ahB = sAh + buf * BM * LDS_STRIDE;
        const __bf16* alB = sAl + buf * BM * LDS_STRIDE;
        const __bf16* bhB = sBh + buf * BN * LDS_STRIDE;
        const __bf16* blB = sBl + buf * BN * LDS_STRIDE;

        v16bf ah[2], al[2];
#pragma unroll
        for (int mi = 0; mi < 2; ++mi) {
            ah[mi] = load_a_frag(ahB, wm + 16 * mi + lane16, hh);
            al[mi] = load_a_frag(alB, wm + 16 * mi + lane16, hh);
        }
#pragma unroll
        for (int ni = 0; ni < 4; ++ni) {
            const v16bf bh = load_b_frag(bhB, wn + 16 * ni + lane16, hh);
            const v16bf bl = load_b_frag(blB, wn + 16 * ni + lane16, hh);
#pragma unroll
            for (int mi = 0; mi < 2; ++mi) {
                // hi*hi + hi*lo + lo*hi  (lo*lo dropped: below fp32 noise floor)
                acc[mi][ni] = __builtin_amdgcn_wmma_f32_16x16x32_bf16(
                    false, ah[mi], false, bh, (short)0, acc[mi][ni], false, false);
                acc[mi][ni] = __builtin_amdgcn_wmma_f32_16x16x32_bf16(
                    false, ah[mi], false, bl, (short)0, acc[mi][ni], false, false);
                acc[mi][ni] = __builtin_amdgcn_wmma_f32_16x16x32_bf16(
                    false, al[mi], false, bh, (short)0, acc[mi][ni], false, false);
            }
        }

        // ---- stage next X tile into the other buffer ----
        if (kt + 1 < KT) {
#pragma unroll
            for (int kk = 0; kk < 4; ++kk) {
                const float f[4] = { xr[kk].x, xr[kk].y, xr[kk].z, xr[kk].w };
                bfx4 h4, l4;
#pragma unroll
                for (int e = 0; e < 4; ++e) {
                    h4[e] = (__bf16)f[e];
                    l4[e] = (__bf16)(f[e] - (float)h4[e]);
                }
                *(bfx4*)&sAh[(nb * BM + kk * 32 + xr_row) * LDS_STRIDE + xr_c4] = h4;
                *(bfx4*)&sAl[(nb * BM + kk * 32 + xr_row) * LDS_STRIDE + xr_c4] = l4;
            }
        }
        __builtin_amdgcn_s_wait_tensorcnt(0);   // TDM for buf^1 done (wave 0)
        __syncthreads();
    }

    // ---- epilogue: add bias, write C (f32 16x16 layout: N=lane16, M=vgpr+8*hh)
#pragma unroll
    for (int mi = 0; mi < 2; ++mi)
#pragma unroll
        for (int ni = 0; ni < 4; ++ni) {
            const int n  = n0 + wn + ni * 16 + lane16;
            const float bv = bias[n];
#pragma unroll
            for (int r = 0; r < 8; ++r) {
                const int m = m0 + wm + mi * 16 + hh * 8 + r;
                out[(size_t)m * D_OUT + n] = acc[mi][ni][r] + bv;
            }
        }
}

// ---------------------------------------------------------------------------
// Host launcher
// Inputs (setup_inputs order): x, weight, bias, V_q, U_T_q, V_scale, U_T_scale
// Workspace use: 64 MB = whi[D_OUT*D_IN] bf16 + wlo[D_OUT*D_IN] bf16
// ---------------------------------------------------------------------------
extern "C" void kernel_launch(void* const* d_in, const int* in_sizes, int n_in,
                              void* d_out, int out_size, void* d_ws, size_t ws_size,
                              hipStream_t stream)
{
    const float* x      = (const float*)d_in[0];
    const float* weight = (const float*)d_in[1];
    const float* bias   = (const float*)d_in[2];
    const int*   Vq     = (const int*)d_in[3];
    const int*   UTq    = (const int*)d_in[4];
    const float* vscale = (const float*)d_in[5];
    const float* uscale = (const float*)d_in[6];
    float*       out    = (float*)d_out;

    __bf16* whi = (__bf16*)d_ws;
    __bf16* wlo = whi + (size_t)D_OUT * D_IN;

    // 80 KB dynamic LDS per block (gfx1250 WGP has 320 KB)
    (void)hipFuncSetAttribute((const void*)lora_gemm_wmma_kernel,
                              hipFuncAttributeMaxDynamicSharedMemorySize,
                              SMEM_BYTES);

    // Pass 1: fold low-rank int8 correction into weight, emit split-bf16 W_eff
    weff_build_kernel<<<dim3(D_OUT / 256, D_IN), 256, 0, stream>>>(
        weight, Vq, UTq, vscale, uscale, whi, wlo);

    // Pass 2: fused GEMM + bias via bf16 WMMA + TDM-fed B tiles
    lora_gemm_wmma_kernel<<<dim3(D_OUT / BN, M_TOTAL / BM), 256, SMEM_BYTES, stream>>>(
        x, whi, wlo, bias, out);
}
